// STPStateMixer_82094004896540
// MI455X (gfx1250) — matrix operand order repaired and checked by reading, not compiled, not run on hardware
//
#include <hip/hip_runtime.h>
#include <math.h>

// Problem sizes (fixed by the reference)
#define BSZ 2
#define LSEQ 1024
#define DMODEL 512
#define DINNER 2048

// Persistent scan kernel partitioning
#define ROWS 32                 // I-rows of F owned per workgroup
#define NBLK (DINNER / ROWS)    // 64 blocks per sample
#define NWG  (BSZ * NBLK)       // 128 workgroups
#define TPB  256                // 8 waves (wave32)
#define ROWS_PER_WAVE 4
#define DPL 16                  // d elements per lane (512 / 32)

typedef _Float16 v8h  __attribute__((ext_vector_type(8)));
typedef _Float16 v16h __attribute__((ext_vector_type(16)));
typedef float    v8f  __attribute__((ext_vector_type(8)));

__device__ __forceinline__ float sigmoidf_(float v) { return 1.0f / (1.0f + __expf(-v)); }
__device__ __forceinline__ float gelu_exact(float v) {
    return 0.5f * v * (1.0f + erff(v * 0.70710678118654752440f));
}

// ---------------------------------------------------------------------------
// Kernel 0: zero the output (harness poisons d_out; we accumulate atomically)
// ---------------------------------------------------------------------------
__global__ void zero_f32(float* __restrict__ p, int n) {
    int i = blockIdx.x * blockDim.x + threadIdx.x;
    if (i < n) p[i] = 0.0f;
}

// ---------------------------------------------------------------------------
// Kernel 1: fp32 -> fp16 conversion (X and W1 operands for the WMMA GEMM)
// ---------------------------------------------------------------------------
__global__ void cvt_f32_to_f16(const float* __restrict__ src, _Float16* __restrict__ dst, int n) {
    int i = blockIdx.x * blockDim.x + threadIdx.x;
    if (i < n) dst[i] = (_Float16)src[i];
}

// ---------------------------------------------------------------------------
// Kernel 2: U = X * W1^T + b1   via v_wmma_f32_16x16x32_f16
//   X  : [BSZ*LSEQ, DMODEL] f16 (row-major)
//   W1 : [DINNER,  DMODEL]  f16 (row-major)  -> B operand column n is W1[n,:]
//   U  : [BSZ*LSEQ, DINNER] f32
// One wave computes one 16x16 output tile; K-loop of 16 WMMAs (K=512).
// ---------------------------------------------------------------------------
__global__ __launch_bounds__(TPB) void wmma_gemm_u(
    const _Float16* __restrict__ Xh,
    const _Float16* __restrict__ W1h,
    const float*    __restrict__ b1,
    float*          __restrict__ U)
{
    const int lane  = threadIdx.x & 31;
    const int wtile = blockIdx.x * (TPB / 32) + (threadIdx.x >> 5);
    const int tm = wtile >> 7;          // 128 N-tiles per M row of tiles
    const int tn = wtile & 127;
    const int m0 = tm * 16, n0 = tn * 16;

    const int row = lane & 15;          // M for A-operand, N for B-operand
    const int ka  = (lane >> 4) * 16;   // A: K base; 16 contiguous K values
    const int kb  = (lane >> 4) * 16;   // B: K base; 16 contiguous K values

    v8f acc = {};
    #pragma unroll 4
    for (int k0 = 0; k0 < DMODEL; k0 += 32) {
        // A operand: lane holds M=row, K = k0 + ka + {0..15} (contiguous)
        const _Float16* ap = Xh + (size_t)(m0 + row) * DMODEL + k0 + ka;
        v8h a_lo = *(const v8h*)(ap);
        v8h a_hi = *(const v8h*)(ap + 8);
        v16h a = __builtin_shufflevector(a_lo, a_hi,
                    0, 1, 2, 3, 4, 5, 6, 7, 8, 9, 10, 11, 12, 13, 14, 15);
        // B operand: lane holds N=row, K = k0 + kb + {0..15} (contiguous)
        const _Float16* bp = W1h + (size_t)(n0 + row) * DMODEL + k0 + kb;
        v16h b = *(const v16h*)(bp);

        acc = __builtin_amdgcn_wmma_f32_16x16x32_f16(
                  /*neg_a=*/false, a, /*neg_b=*/false, b,
                  /*c_mod=*/(short)0, acc, /*reuse_a=*/false, /*reuse_b=*/false);
    }

    // C/D layout: element r -> M = m0 + (lane>>4)*8 + r, N = n0 + (lane&15)
    const int n     = n0 + (lane & 15);
    const int mbase = m0 + ((lane >> 4) << 3);
    const float bias = b1[n];
    #pragma unroll
    for (int r = 0; r < 8; ++r)
        U[(size_t)(mbase + r) * DINNER + n] = acc[r] + bias;
}

// ---------------------------------------------------------------------------
// Kernel 3: persistent fast-weight scan.
// Grid = 128 WGs; WG g owns sample b = g/64 and rows [i0, i0+32) of F.
// F, R=1-sigmoid(Lambda), G=sigmoid(Gamma) live entirely in VGPRs
// (lane owns d = lane + 32*j, j=0..15; wave owns 4 rows).
// W2[:, i0:i0+32] is staged once in LDS; x_t and g shared via LDS per step.
// Cross-WG combine of the output projection is done with global f32 atomics
// (no grid-wide barrier ever needed).
// ---------------------------------------------------------------------------
__global__ __launch_bounds__(TPB, 1) void fastweight_scan(
    const float* __restrict__ x,     // [B, L, D]
    const float* __restrict__ W2,    // [D, I]
    const float* __restrict__ b2,    // [D]
    const float* __restrict__ LamF,  // [I, D]
    const float* __restrict__ GamF,  // [I, D]
    const float* __restrict__ U,     // [B*L, I] precomputed X*W1^T + b1
    float*       __restrict__ out)   // [B, L, D]
{
    extern __shared__ float lds[];
    float* W2blk = lds;                       // [ROWS][DMODEL] = 32*512 floats
    float* x_lds = lds + ROWS * DMODEL;       // [DMODEL]
    float* g_lds = x_lds + DMODEL;            // [ROWS]

    const int tid  = threadIdx.x;
    const int wave = tid >> 5;
    const int lane = tid & 31;
    const int g    = blockIdx.x;
    const int b    = g >> 6;                  // g / NBLK
    const int blk  = g & (NBLK - 1);
    const int i0   = blk * ROWS;

    // Stage the time-invariant W2 slab: W2blk[i][d] = W2[d][i0+i]
    for (int idx = tid; idx < ROWS * DMODEL; idx += TPB) {
        const int i = idx & (ROWS - 1);
        const int d = idx >> 5;
        W2blk[i * DMODEL + d] = W2[(size_t)d * DINNER + i0 + i];
    }

    // Register-resident state + decay/gain tiles
    float F[ROWS_PER_WAVE][DPL], R[ROWS_PER_WAVE][DPL], G[ROWS_PER_WAVE][DPL];
    #pragma unroll
    for (int r = 0; r < ROWS_PER_WAVE; ++r) {
        const int ig = i0 + wave * ROWS_PER_WAVE + r;
        #pragma unroll
        for (int j = 0; j < DPL; ++j) {
            const int d = lane + 32 * j;
            R[r][j] = 1.0f - sigmoidf_(LamF[(size_t)ig * DMODEL + d]);
            G[r][j] = sigmoidf_(GamF[(size_t)ig * DMODEL + d]);
            F[r][j] = 0.0f;
        }
    }
    __syncthreads();

    const float* xb   = x   + (size_t)b * LSEQ * DMODEL;
    const float* Ub   = U   + (size_t)b * LSEQ * DINNER;
    float*       outb = out + (size_t)b * LSEQ * DMODEL;

    for (int t = 0; t < LSEQ; ++t) {
        // Stage x_t into LDS (128 lanes x float4)
        if (tid < DMODEL / 4)
            ((float4*)x_lds)[tid] = ((const float4*)(xb + (size_t)t * DMODEL))[tid];
        if (t + 1 < LSEQ)
            __builtin_prefetch(xb + (size_t)(t + 1) * DMODEL, 0, 1);
        __syncthreads();

        float xr[DPL];
        #pragma unroll
        for (int j = 0; j < DPL; ++j) xr[j] = x_lds[lane + 32 * j];

        // h_pre for this wave's 4 rows: register matvec + butterfly reduce + U
        float hp[ROWS_PER_WAVE];
        #pragma unroll
        for (int r = 0; r < ROWS_PER_WAVE; ++r) {
            float s = 0.0f;
            #pragma unroll
            for (int j = 0; j < DPL; ++j) s = fmaf(F[r][j], xr[j], s);
            #pragma unroll
            for (int off = 16; off >= 1; off >>= 1) s += __shfl_xor(s, off, 32);
            s += Ub[(size_t)t * DINNER + i0 + wave * ROWS_PER_WAVE + r]; // W1 x_t + b1
            hp[r] = s;
            const float gg = gelu_exact(s);
            if (lane == r) g_lds[wave * ROWS_PER_WAVE + r] = gg;
        }
        __syncthreads();

        // Hebbian update (purely register-local): F = R*F + G * h_pre * x
        #pragma unroll
        for (int r = 0; r < ROWS_PER_WAVE; ++r) {
            const float hx = hp[r];
            #pragma unroll
            for (int j = 0; j < DPL; ++j)
                F[r][j] = fmaf(R[r][j], F[r][j], G[r][j] * hx * xr[j]);
        }

        // Output projection partial over this WG's 32 rows; atomically combined
        for (int dd = tid; dd < DMODEL; dd += TPB) {
            float acc = 0.0f;
            #pragma unroll 8
            for (int i = 0; i < ROWS; ++i)
                acc = fmaf(W2blk[i * DMODEL + dd], g_lds[i], acc);
            if (blk == 0) acc += b2[dd];   // bias added exactly once per (b,t,d)
            atomicAdd(&outb[(size_t)t * DMODEL + dd], acc);
        }
        __syncthreads();   // protect x_lds/g_lds before next step overwrites
    }
}

// ---------------------------------------------------------------------------
// Host-side launcher
// ---------------------------------------------------------------------------
extern "C" void kernel_launch(void* const* d_in, const int* in_sizes, int n_in,
                              void* d_out, int out_size, void* d_ws, size_t ws_size,
                              hipStream_t stream) {
    (void)in_sizes; (void)n_in; (void)out_size; (void)ws_size;

    const float* x    = (const float*)d_in[0];   // [B, L, D]
    const float* W1   = (const float*)d_in[1];   // [I, D]
    const float* b1   = (const float*)d_in[2];   // [I]
    const float* W2   = (const float*)d_in[3];   // [D, I]
    const float* b2   = (const float*)d_in[4];   // [D]
    const float* LamF = (const float*)d_in[5];   // [I, D]
    const float* GamF = (const float*)d_in[6];   // [I, D]
    float* out        = (float*)d_out;           // [B, L, D]

    // Workspace layout
    char* ws = (char*)d_ws;
    float*     U   = (float*)ws;                                   // 16 MB
    _Float16*  Xh  = (_Float16*)(ws + (size_t)16 * 1024 * 1024);   //  2 MB
    _Float16*  W1h = (_Float16*)(ws + (size_t)18 * 1024 * 1024);   //  2 MB

    const int n_out = BSZ * LSEQ * DMODEL;           // 1,048,576
    const int n_x   = BSZ * LSEQ * DMODEL;           // 1,048,576
    const int n_w1  = DINNER * DMODEL;               // 1,048,576

    // 0) zero output (we accumulate with atomics)
    zero_f32<<<(n_out + 255) / 256, 256, 0, stream>>>(out, n_out);

    // 1) fp16 operands for the WMMA GEMM
    cvt_f32_to_f16<<<(n_x  + 255) / 256, 256, 0, stream>>>(x,  Xh,  n_x);
    cvt_f32_to_f16<<<(n_w1 + 255) / 256, 256, 0, stream>>>(W1, W1h, n_w1);

    // 2) U = X * W1^T + b1  (M=N=2048, K=512) — 16384 16x16 tiles, 8 waves/WG
    wmma_gemm_u<<<16384 / (TPB / 32), TPB, 0, stream>>>(Xh, W1h, b1, U);

    // 3) persistent scan, 128 WGs, dynamic LDS = 32*512 + 512 + 32 floats
    const int lds_bytes = (ROWS * DMODEL + DMODEL + ROWS) * (int)sizeof(float);
    (void)hipFuncSetAttribute((const void*)fastweight_scan,
                              hipFuncAttributeMaxDynamicSharedMemorySize, lds_bytes);
    fastweight_scan<<<NWG, TPB, lds_bytes, stream>>>(x, W2, b2, LamF, GamF, U, out);
}
